// ResidualVectorQuantization_4982162063513
// MI455X (gfx1250) — matrix operand order repaired and checked by reading, not compiled, not run on hardware
//
#include <hip/hip_runtime.h>
#include <stdint.h>

// Problem constants (from reference)
#define NQ      8        // num quantizers
#define KCODES  1024     // codebook entries
#define DIM     256      // embedding dim
#define NTOK    65536    // 16 * 4096 tokens
#define WAVES   8        // waves per block in stage kernel
#define BLOCKS_STAGE 512 // NTOK / (WAVES * 16)
#define CHUNK   64       // codebook rows staged in LDS per step
#define NCHUNK  (KCODES / CHUNK)          // 16
#define ROWP    (DIM + 8)                 // padded LDS row (ushorts): 528B = 132 dw
#define CHUNKW  (CHUNK * ROWP)            // ushorts per chunk buffer
#define EPSN    1e-12f

#if defined(__has_builtin)
#if __has_builtin(__builtin_amdgcn_global_load_async_to_lds_b128) && \
    __has_builtin(__builtin_amdgcn_s_wait_asynccnt)
#define HAVE_ASYNC_LDS 1
#endif
#endif
#ifndef HAVE_ASYNC_LDS
#define HAVE_ASYNC_LDS 0
#endif

typedef __attribute__((ext_vector_type(16))) __bf16 v16bf;
typedef __attribute__((ext_vector_type(8)))  float  v8f;
typedef int v4i_vs __attribute__((vector_size(4 * sizeof(int))));

union AB16 { uint4 q[2]; v16bf v; };

__device__ __forceinline__ float wave_sum32(float v) {
  v += __shfl_xor(v, 16, 32);
  v += __shfl_xor(v, 8, 32);
  v += __shfl_xor(v, 4, 32);
  v += __shfl_xor(v, 2, 32);
  v += __shfl_xor(v, 1, 32);
  return v;
}

__device__ __forceinline__ unsigned short bf16rne(float f) {
  union { float f; uint32_t u; } c; c.f = f;
  uint32_t u = c.u;
  u += 0x7FFFu + ((u >> 16) & 1u);           // round-to-nearest-even
  return (unsigned short)(u >> 16);
}

// Cooperative stage of one 64-code chunk (64 x 256 bf16, padded rows) into LDS.
// 256 threads: thread -> (row = tid>>2, quarter = tid&3), 8 x b128 each.
__device__ __forceinline__ void chunk_load(unsigned short* __restrict__ dstLds,
                                           const unsigned short* __restrict__ srcG,
                                           int tid) {
  const int row = tid >> 2;
  const int q   = tid & 3;
  const unsigned short* g = srcG + (size_t)row * DIM + q * 64;   // 128B segment
  unsigned short* l = dstLds + row * ROWP + q * 64;
#if HAVE_ASYNC_LDS
#pragma unroll
  for (int i = 0; i < 8; ++i)
    __builtin_amdgcn_global_load_async_to_lds_b128(
        (v4i_vs*)(g + i * 8), (v4i_vs*)(l + i * 8), 0, 0);
#else
#pragma unroll
  for (int i = 0; i < 8; ++i)
    *(uint4*)(l + i * 8) = *(const uint4*)(g + i * 8);
#endif
}

// ---------------------------------------------------------------- init -----
__global__ void rvq_init(float* __restrict__ used) {
  int i = blockIdx.x * 256 + threadIdx.x;
  if (i < NQ * KCODES) used[i] = 0.0f;
}

// ------------------------------------------------- codebook prep (8192 rows)
__global__ __launch_bounds__(256) void rvq_prep(const float* __restrict__ cb,
                                                unsigned short* __restrict__ enbf,
                                                float* __restrict__ invcb) {
  int row  = blockIdx.x * 8 + (threadIdx.x >> 5);   // one wave per row
  int lane = threadIdx.x & 31;
  const float* p = cb + (size_t)row * DIM + lane * 8;
  float4 a = *(const float4*)p;
  float4 b = *(const float4*)(p + 4);
  float ss = a.x*a.x + a.y*a.y + a.z*a.z + a.w*a.w +
             b.x*b.x + b.y*b.y + b.z*b.z + b.w*b.w;
  ss = wave_sum32(ss);
  float inv = 1.0f / fmaxf(sqrtf(ss), EPSN);
  if (lane == 0) invcb[row] = inv;
  uint32_t w0 = (uint32_t)bf16rne(a.x*inv) | ((uint32_t)bf16rne(a.y*inv) << 16);
  uint32_t w1 = (uint32_t)bf16rne(a.z*inv) | ((uint32_t)bf16rne(a.w*inv) << 16);
  uint32_t w2 = (uint32_t)bf16rne(b.x*inv) | ((uint32_t)bf16rne(b.y*inv) << 16);
  uint32_t w3 = (uint32_t)bf16rne(b.z*inv) | ((uint32_t)bf16rne(b.w*inv) << 16);
  *(uint4*)(enbf + (size_t)row * DIM + lane * 8) = make_uint4(w0, w1, w2, w3);
}

// ------------------------------------------------------------ stage kernel -
// 8 waves x 16 tokens. A fragments in registers; codebook chunks staged into
// LDS with double-buffered async global->LDS copies shared by all 8 waves.
__global__ __launch_bounds__(WAVES * 32) void rvq_stage(
    const float* __restrict__ cb,            // [NQ][KCODES][DIM] fp32
    const unsigned short* __restrict__ enbf, // [NQ][KCODES][DIM] bf16 normalized
    const float* __restrict__ invcb,         // [NQ*KCODES]
    float* __restrict__ res,                 // [NTOK][DIM] residual (in/out)
    float* __restrict__ qout,                // [NTOK][DIM] accumulated q
    float* __restrict__ idx_out,             // [NQ][NTOK] (as float)
    float* __restrict__ used,                // [NQ][KCODES]
    float* __restrict__ part,                // [NQ][BLOCKS][WAVES][2] loss partials
    int s) {
  // One pool: first used as zn staging (8 waves x 16 x 256 bf16 = 32768 us),
  // then reused as two padded codebook chunk buffers (2 x CHUNKW).
  __shared__ unsigned short sBuf[2 * CHUNKW];   // 67584 B

  const int tid  = threadIdx.x;
  const int wv   = tid >> 5;
  const int lane = tid & 31;
  const int m    = lane & 15;     // column / row-in-tile residue
  const int half = lane >> 4;     // K sub-block / M half selector
  const int t0   = (blockIdx.x * WAVES + wv) * 16;
  unsigned short* zn = sBuf + wv * 16 * DIM;

  // ---- phase 1: row norms + bf16 zn tile into LDS --------------------------
  float invreg[16];
#pragma unroll
  for (int r = 0; r < 16; ++r) {
    const float* rp = res + (size_t)(t0 + r) * DIM + lane * 8;
    float4 a = *(const float4*)rp;
    float4 b = *(const float4*)(rp + 4);
    float ss = a.x*a.x + a.y*a.y + a.z*a.z + a.w*a.w +
               b.x*b.x + b.y*b.y + b.z*b.z + b.w*b.w;
    ss = wave_sum32(ss);
    float inv = 1.0f / fmaxf(sqrtf(ss), EPSN);
    invreg[r] = inv;
    uint32_t w0 = (uint32_t)bf16rne(a.x*inv) | ((uint32_t)bf16rne(a.y*inv) << 16);
    uint32_t w1 = (uint32_t)bf16rne(a.z*inv) | ((uint32_t)bf16rne(a.w*inv) << 16);
    uint32_t w2 = (uint32_t)bf16rne(b.x*inv) | ((uint32_t)bf16rne(b.y*inv) << 16);
    uint32_t w3 = (uint32_t)bf16rne(b.z*inv) | ((uint32_t)bf16rne(b.w*inv) << 16);
    *(uint4*)(zn + r * DIM + lane * 8) = make_uint4(w0, w1, w2, w3);
  }

  // ---- phase 2: A fragments (16x32 bf16 per k-chunk, ISA lane layout) ------
  AB16 afrag[8];
#pragma unroll
  for (int kc = 0; kc < 8; ++kc) {
    afrag[kc].q[0] = *(const uint4*)(zn + m * DIM + kc * 32 + half * 8);
    afrag[kc].q[1] = *(const uint4*)(zn + m * DIM + kc * 32 + 16 + half * 8);
  }
  __syncthreads();   // zn region now free for chunk buffers

  // ---- phase 3: double-buffered async codebook staging + GEMM + argmax -----
  float bestv[8];
  int   besti[8];
#pragma unroll
  for (int i = 0; i < 8; ++i) { bestv[i] = -3.0e38f; besti[i] = 0; }

  const unsigned short* enb = enbf + ((size_t)s << 18);  // s*KCODES*DIM
  chunk_load(sBuf, enb, tid);                             // chunk 0 -> buf0

  for (int c = 0; c < NCHUNK; ++c) {
    unsigned short* cur = sBuf + (c & 1) * CHUNKW;
    if (c + 1 < NCHUNK) {
      chunk_load(sBuf + ((c + 1) & 1) * CHUNKW, enb + (size_t)(c + 1) * CHUNK * DIM, tid);
#if HAVE_ASYNC_LDS
      __builtin_amdgcn_s_wait_asynccnt(8);   // chunk c landed (in-order), c+1 in flight
#endif
    } else {
#if HAVE_ASYNC_LDS
      __builtin_amdgcn_s_wait_asynccnt(0);   // last chunk fully landed
#endif
    }
    __syncthreads();   // chunk c visible to all waves

#pragma unroll
    for (int tile = 0; tile < CHUNK / 16; ++tile) {
      const int codeLocal = tile * 16 + m;
      const int code = c * CHUNK + codeLocal;
      // B lane layout: lane m, half h holds K = h*8 + 0..7 and 16 + h*8 + 0..7 of row `code`
      const unsigned short* bp = cur + codeLocal * ROWP + half * 8;
      v8f cc = {};
#pragma unroll
      for (int kc = 0; kc < 8; ++kc) {
        AB16 bfrag;
        bfrag.q[0] = *(const uint4*)(bp + kc * 32);
        bfrag.q[1] = *(const uint4*)(bp + kc * 32 + 16);
        cc = __builtin_amdgcn_wmma_f32_16x16x32_bf16(
            /*neg_a=*/false, afrag[kc].v, /*neg_b=*/false, bfrag.v,
            /*c_mod=*/(short)0, cc, /*reuse_a=*/false, /*reuse_b=*/false);
      }
#pragma unroll
      for (int i = 0; i < 8; ++i) {
        if (cc[i] > bestv[i]) { bestv[i] = cc[i]; besti[i] = code; }
      }
    }
    __syncthreads();   // done with buf before it is refilled
  }

  // ---- phase 4: argmax across the 16 lanes of each half (tie -> lower idx) -
#pragma unroll
  for (int msk = 1; msk < 16; msk <<= 1) {
#pragma unroll
    for (int i = 0; i < 8; ++i) {
      float ov = __shfl_xor(bestv[i], msk, 32);
      int   oi = __shfl_xor(besti[i], msk, 32);
      if (ov > bestv[i] || (ov == bestv[i] && oi < besti[i])) {
        bestv[i] = ov; besti[i] = oi;
      }
    }
  }
  if (m == 0) {  // lanes 0 (rows 0-7) and 16 (rows 8-15)
    const int rbase = half * 8;
#pragma unroll
    for (int i = 0; i < 8; ++i) {
      idx_out[(size_t)s * NTOK + t0 + rbase + i] = (float)besti[i];
      used[s * KCODES + besti[i]] = 1.0f;   // benign races: same value
    }
  }

  // ---- phase 5: residual update, qout accumulate, loss sums ----------------
  float s1 = 0.0f, s2 = 0.0f;
#pragma unroll
  for (int r = 0; r < 16; ++r) {
    const int src  = (r < 8) ? 0 : 16;
    const int slot = r & 7;
    const int idx  = __shfl(besti[slot], src, 32);
    const float invz = invreg[r];
    const float invc = invcb[s * KCODES + idx];
    const float* cbr = cb + ((size_t)(s * KCODES + idx)) * DIM + lane * 8;
    float* rp = res  + (size_t)(t0 + r) * DIM + lane * 8;
    float* qp = qout + (size_t)(t0 + r) * DIM + lane * 8;
    float4 r0 = *(const float4*)rp, r1 = *(const float4*)(rp + 4);
    float4 c0 = *(const float4*)cbr, c1 = *(const float4*)(cbr + 4);
    float rv[8] = {r0.x, r0.y, r0.z, r0.w, r1.x, r1.y, r1.z, r1.w};
    float cv[8] = {c0.x, c0.y, c0.z, c0.w, c1.x, c1.y, c1.z, c1.w};
    float nr[8], zq[8];
#pragma unroll
    for (int e = 0; e < 8; ++e) {
      float q  = cv[e] * invc;                 // zq (forward value of STE q)
      float z  = rv[e] * invz;                 // zn
      float d1 = q - z;            s1 += d1 * d1;   // commit term
      float d2 = 2.0f * q - rv[e]; s2 += d2 * d2;   // (q - new_res)^2
      nr[e] = rv[e] - q;
      zq[e] = q;
    }
    *(float4*)rp       = make_float4(nr[0], nr[1], nr[2], nr[3]);
    *((float4*)rp + 1) = make_float4(nr[4], nr[5], nr[6], nr[7]);
    if (s == 0) {
      *(float4*)qp       = make_float4(zq[0], zq[1], zq[2], zq[3]);
      *((float4*)qp + 1) = make_float4(zq[4], zq[5], zq[6], zq[7]);
    } else {
      float4 q0 = *(const float4*)qp, q1 = *((const float4*)qp + 1);
      *(float4*)qp       = make_float4(q0.x + zq[0], q0.y + zq[1], q0.z + zq[2], q0.w + zq[3]);
      *((float4*)qp + 1) = make_float4(q1.x + zq[4], q1.y + zq[5], q1.z + zq[6], q1.w + zq[7]);
    }
  }

  // ---- phase 6: per-wave deterministic loss partials -----------------------
  s1 = wave_sum32(s1);
  s2 = wave_sum32(s2);
  if (lane == 0) {
    size_t pi = (((size_t)s * BLOCKS_STAGE + blockIdx.x) * WAVES + wv) * 2;
    part[pi]     = s1;
    part[pi + 1] = s2;
  }
}

// ------------------------------------------------------------- finalize ----
__global__ __launch_bounds__(256) void rvq_fin(const float* __restrict__ used,
                                               const float* __restrict__ part,
                                               float* __restrict__ loss_out,
                                               float* __restrict__ usage_out) {
  __shared__ float red[256];
  const int tid = threadIdx.x;
  for (int s = 0; s < NQ; ++s) {
    float p = 0.0f;
    for (int i = tid; i < KCODES; i += 256) p += used[s * KCODES + i];
    red[tid] = p; __syncthreads();
    for (int k = 128; k > 0; k >>= 1) {
      if (tid < k) red[tid] += red[tid + k];
      __syncthreads();
    }
    if (tid == 0) usage_out[s] = red[0] / (float)KCODES;
    __syncthreads();
  }
  float a = 0.0f;
  for (int i = tid; i < NQ * BLOCKS_STAGE * WAVES; i += 256)
    a += 0.25f * part[i * 2] + 0.4f * part[i * 2 + 1];
  red[tid] = a; __syncthreads();
  for (int k = 128; k > 0; k >>= 1) {
    if (tid < k) red[tid] += red[tid + k];
    __syncthreads();
  }
  if (tid == 0)
    loss_out[0] = red[0] / ((float)NQ * (float)NTOK * (float)DIM);
}

// ------------------------------------------------------------------ launch -
extern "C" void kernel_launch(void* const* d_in, const int* in_sizes, int n_in,
                              void* d_out, int out_size, void* d_ws, size_t ws_size,
                              hipStream_t stream) {
  const float* x  = (const float*)d_in[0];   // [16,4096,256]
  const float* cb = (const float*)d_in[1];   // [8,1024,256]

  float* out       = (float*)d_out;
  float* qout      = out;                                   // 16777216
  float* idx_out   = out + (size_t)NTOK * DIM;              //   524288
  float* loss_out  = idx_out + (size_t)NQ * NTOK;           //        1
  float* usage_out = loss_out + 1;                          //        8

  char* w = (char*)d_ws;
  size_t off = 0;
  float* res = (float*)(w + off);            off += (size_t)NTOK * DIM * sizeof(float); // 64 MB
  unsigned short* enbf = (unsigned short*)(w + off); off += (size_t)NQ * KCODES * DIM * 2; // 4 MB
  float* invcb = (float*)(w + off);          off += (size_t)NQ * KCODES * sizeof(float);
  float* used  = (float*)(w + off);          off += (size_t)NQ * KCODES * sizeof(float);
  float* part  = (float*)(w + off);          // NQ*BLOCKS_STAGE*WAVES*2 floats (256 KB)

  // residual starts as a copy of x (never mutate inputs)
  (void)hipMemcpyAsync(res, x, (size_t)NTOK * DIM * sizeof(float),
                       hipMemcpyDeviceToDevice, stream);

  rvq_init<<<(NQ * KCODES + 255) / 256, 256, 0, stream>>>(used);
  rvq_prep<<<NQ * KCODES / 8, 256, 0, stream>>>(cb, enbf, invcb);

  for (int s = 0; s < NQ; ++s)
    rvq_stage<<<BLOCKS_STAGE, WAVES * 32, 0, stream>>>(
        cb, enbf, invcb, res, qout, idx_out, used, part, s);

  rvq_fin<<<1, 256, 0, stream>>>(used, part, loss_out, usage_out);
}
